// QRecurrentSNN_83468394430849
// MI455X (gfx1250) — compile-verified, matrix operand order checked by reading, not run on hardware
//
#include <hip/hip_runtime.h>
#include <hip/hip_bf16.h>

// ---------------------------------------------------------------------------
// QRecurrentSNN on gfx1250 (MI455X)
//   HIDDEN=1024 (512 reg + 512 adaptive), INPUT=256, BATCH=32, T=1000, OUT=20
//   Per step: i_t = [x_t | z] @ W^T  via v_wmma_f32_16x16x32_bf16 with TWO
//   independent accumulator chains (kills WMMA hazard NOPs), fused LIF/ALIF
//   elementwise update in the WMMA epilogue.
// ---------------------------------------------------------------------------

#define SNN_B      32
#define SNN_T      1000
#define SNN_I      256
#define SNN_H      1024
#define SNN_K      (SNN_I + SNN_H)   // 1280
#define SNN_OUT    20

#define SNN_ALPHA  0.95122942450071400910f   // exp(-1/20)
#define SNN_RHO    0.99501247919268231110f   // exp(-1/200)
#define SNN_DTRACE 0.95122942450071400910f   // exp(-1/20)
#define SNN_THR    0.6f
#define SNN_BETA   0.0174f                   // 1.74 * 0.01

typedef __attribute__((ext_vector_type(16))) __bf16 v16bf;
typedef __attribute__((ext_vector_type(8)))  __bf16 bf16x8;
typedef __attribute__((ext_vector_type(8)))  float  v8f;

__device__ __forceinline__ bf16x8 cvt8_f32_bf16(float4 f0, float4 f1) {
    bf16x8 r;
    r[0] = (__bf16)f0.x; r[1] = (__bf16)f0.y; r[2] = (__bf16)f0.z; r[3] = (__bf16)f0.w;
    r[4] = (__bf16)f1.x; r[5] = (__bf16)f1.y; r[6] = (__bf16)f1.z; r[7] = (__bf16)f1.w;
    return r;
}

__device__ __forceinline__ v16bf cat16(bf16x8 lo, bf16x8 hi) {
    return __builtin_shufflevector(lo, hi, 0,1,2,3,4,5,6,7,8,9,10,11,12,13,14,15);
}

// ---------------------------------------------------------------------------
// Prep: W_bf16[h][k] = bf16( w0*C0 + w1*C1 )    (2.6 MB, L2-resident all run)
// ---------------------------------------------------------------------------
__global__ void snn_build_w(const float* __restrict__ C0,
                            const float* __restrict__ C1,
                            const float* __restrict__ bw,
                            __bf16* __restrict__ Wq) {
    size_t i = (size_t)blockIdx.x * blockDim.x + threadIdx.x;
    if (i < (size_t)SNN_H * SNN_K) {
        float w = bw[0] * C0[i] + bw[1] * C1[i];
        Wq[i] = (__bf16)w;
    }
}

// Optional prep: pre-quantize the whole input tensor x -> bf16 (16 MB).
__global__ void snn_build_xq(const float* __restrict__ x,
                             __bf16* __restrict__ xq) {
    size_t i = (size_t)blockIdx.x * blockDim.x + threadIdx.x;
    if (i < (size_t)SNN_B * SNN_T * SNN_I) xq[i] = (__bf16)x[i];
}

// Zero persistent state (v, a, trace f32; both z bf16 buffers)
__global__ void snn_init_state(float* __restrict__ v, float* __restrict__ a,
                               float* __restrict__ tr, __bf16* __restrict__ zbuf) {
    int i = blockIdx.x * blockDim.x + threadIdx.x;
    if (i < SNN_B * SNN_H) {
        v[i] = 0.f; a[i] = 0.f; tr[i] = 0.f;
        zbuf[i] = (__bf16)0.f;
        zbuf[SNN_B * SNN_H + i] = (__bf16)0.f;
    }
}

// ---------------------------------------------------------------------------
// One recurrent timestep. grid = 64 blocks (N-tiles) x 64 threads (2 waves =
// 2 M-tiles). Each wave: one 16x16 tile of i_t, K=1280 as 40 bf16 WMMAs split
// over two accumulator chains, fused LIF/ALIF epilogue on the accumulators.
// ---------------------------------------------------------------------------
template <bool USE_XQ>
__global__ void __launch_bounds__(64)
snn_step(const float*  __restrict__ x,      // [32][1000][256] f32
         const __bf16* __restrict__ xq,     // [32][1000][256] bf16 (if USE_XQ)
         const __bf16* __restrict__ Wq,     // [1024][1280]
         __bf16*       __restrict__ zbuf,   // [2][32][1024]
         float* __restrict__ v, float* __restrict__ a, float* __restrict__ tr,
         float* __restrict__ spikes_out,    // d_out + 640         [32][1000][1024]
         float* __restrict__ trace_out,     // d_out + 640 + 32.7M [32][1000][1024]
         int t)
{
    const int lane  = threadIdx.x & 31;
    const int mtile = threadIdx.x >> 5;     // 0..1  (batch tile)
    const int ntile = blockIdx.x;           // 0..63 (hidden tile)
    const int l15   = lane & 15;
    const int kh    = lane >> 4;            // half-wave K split

    const __bf16* zin  = zbuf + (size_t)(t & 1) * (SNN_B * SNN_H);
    __bf16*       zout = zbuf + (size_t)((t + 1) & 1) * (SNN_B * SNN_H);

    // A fragment source rows (lane = M row), B fragment source rows (lane = N col)
    const int bRow = mtile * 16 + l15;                    // batch index for A loads
    const int hCol = ntile * 16 + l15;                    // hidden index for B loads
    const float*  xrow  = x  + ((size_t)bRow * SNN_T + t) * SNN_I;
    const __bf16* xqrow = xq + ((size_t)bRow * SNN_T + t) * SNN_I;
    const __bf16* zrow  = zin + (size_t)bRow * SNN_H;
    const __bf16* wrow  = Wq + (size_t)hCol * SNN_K;

    // Two independent accumulator chains -> no WMMA->WMMA serial dependence,
    // scheduler interleaves loads/converts of one chain under the other.
    v8f acc0 = {};
    v8f acc1 = {};

    // ---- K chunks 0..255 : x_t ----
#pragma unroll
    for (int c = 0; c < SNN_I / 32; ++c) {
        const int k0 = c * 32;
        bf16x8 alo, ahi;
        if (USE_XQ) {
            const __bf16* xp = xqrow + k0 + kh * 8;
            alo = *(const bf16x8*)xp;
            ahi = *(const bf16x8*)(xp + 16);
        } else {
            const float* xp = xrow + k0 + kh * 8;
            alo = cvt8_f32_bf16(*(const float4*)xp,        *(const float4*)(xp + 4));
            ahi = cvt8_f32_bf16(*(const float4*)(xp + 16), *(const float4*)(xp + 20));
        }
        const __bf16* wp = wrow + k0 + kh * 8;
        bf16x8 blo = *(const bf16x8*)wp;
        bf16x8 bhi = *(const bf16x8*)(wp + 16);
        if (c & 1)
            acc1 = __builtin_amdgcn_wmma_f32_16x16x32_bf16(
                false, cat16(alo, ahi), false, cat16(blo, bhi),
                (short)0, acc1, false, false);
        else
            acc0 = __builtin_amdgcn_wmma_f32_16x16x32_bf16(
                false, cat16(alo, ahi), false, cat16(blo, bhi),
                (short)0, acc0, false, false);
    }

    // ---- K chunks 256..1279 : recurrent spikes (already bf16, 0/1 exact) ----
#pragma unroll
    for (int c = 0; c < SNN_H / 32; ++c) {
        const int k0 = SNN_I + c * 32;
        const __bf16* zp = zrow + (k0 - SNN_I) + kh * 8;
        bf16x8 alo = *(const bf16x8*)zp;
        bf16x8 ahi = *(const bf16x8*)(zp + 16);
        const __bf16* wp = wrow + k0 + kh * 8;
        bf16x8 blo = *(const bf16x8*)wp;
        bf16x8 bhi = *(const bf16x8*)(wp + 16);
        if (c & 1)
            acc1 = __builtin_amdgcn_wmma_f32_16x16x32_bf16(
                false, cat16(alo, ahi), false, cat16(blo, bhi),
                (short)0, acc1, false, false);
        else
            acc0 = __builtin_amdgcn_wmma_f32_16x16x32_bf16(
                false, cat16(alo, ahi), false, cat16(blo, bhi),
                (short)0, acc0, false, false);
    }

    // ---- fused LIF/ALIF epilogue ----
    // C/D layout: VGPR j, lanes 0-15 -> M=j ; lanes 16-31 -> M=j+8 ; N = lane&15
    const float beta = (ntile >= 32) ? SNN_BETA : 0.f;   // hCol>=512 => adaptive
#pragma unroll
    for (int j = 0; j < 8; ++j) {
        const int bOut = mtile * 16 + j + 8 * kh;
        const int s    = bOut * SNN_H + hCol;

        float vv = v[s], aa = a[s], tt = tr[s];
        float zprev = (float)zin[s];
        float i_t = acc0[j] + acc1[j];

        float thr_t  = SNN_THR + beta * aa;
        float v_new  = SNN_ALPHA * vv + i_t - zprev * SNN_THR;
        float z_new  = (v_new - thr_t) > 0.f ? 1.f : 0.f;   // spike fwd
        float a_new  = SNN_RHO * aa + z_new;
        float t_new  = SNN_DTRACE * tt + (1.f - SNN_DTRACE) * z_new;

        v[s] = v_new; a[s] = a_new; tr[s] = t_new;
        zout[s] = (__bf16)z_new;

        const size_t seq = (size_t)bOut * (SNN_T * SNN_H) + (size_t)t * SNN_H + hCol;
        spikes_out[seq] = z_new;
        trace_out[seq]  = t_new;
    }
}

// ---------------------------------------------------------------------------
// Readout GEMV (32x20, tiny) + emit zf, vf, af
// d_out layout (floats): [0,640) output | [640,..) spikes_seq | trace_seq |
//                        zf | vf | af
// ---------------------------------------------------------------------------
#define OFF_SPIKES  640
#define OFF_TRACES  (640 + SNN_B * SNN_T * SNN_H)
#define OFF_ZF      (640 + 2 * SNN_B * SNN_T * SNN_H)
#define OFF_VF      (OFF_ZF + SNN_B * SNN_H)
#define OFF_AF      (OFF_VF + SNN_B * SNN_H)

__global__ void snn_finalize(const float* __restrict__ tr,
                             const float* __restrict__ v,
                             const float* __restrict__ a,
                             const __bf16* __restrict__ zfinal,
                             const float* __restrict__ W_out,
                             const float* __restrict__ b_out,
                             float* __restrict__ out)
{
    if (blockIdx.x == 0) {
        for (int tid = threadIdx.x; tid < SNN_B * SNN_OUT; tid += blockDim.x) {
            const int b = tid / SNN_OUT, o = tid % SNN_OUT;
            float s = b_out[o];
            const float* trb = tr + b * SNN_H;
            const float* wo  = W_out + o * SNN_H;
            for (int h = 0; h < SNN_H; ++h) s = fmaf(trb[h], wo[h], s);
            out[tid] = s;
        }
    } else {
        const int i = (blockIdx.x - 1) * blockDim.x + threadIdx.x;
        if (i < SNN_B * SNN_H) {
            out[OFF_ZF + i] = (float)zfinal[i];
            out[OFF_VF + i] = v[i];
            out[OFF_AF + i] = a[i];
        }
    }
}

// ---------------------------------------------------------------------------
extern "C" void kernel_launch(void* const* d_in, const int* in_sizes, int n_in,
                              void* d_out, int out_size, void* d_ws, size_t ws_size,
                              hipStream_t stream) {
    const float* x     = (const float*)d_in[0];   // [32][1000][256]
    const float* C0    = (const float*)d_in[1];   // [1024][1280]
    const float* C1    = (const float*)d_in[2];   // [1024][1280]
    const float* bw    = (const float*)d_in[3];   // [2]
    const float* W_out = (const float*)d_in[4];   // [20][1024]
    const float* b_out = (const float*)d_in[5];   // [20]
    float* out = (float*)d_out;

    // Workspace layout (base ~3.15 MB; +16 MB if xq fits)
    const size_t szWq   = (size_t)SNN_H * SNN_K * 2;          // 2,621,440
    const size_t szZbuf = (size_t)2 * SNN_B * SNN_H * 2;      //   131,072
    const size_t szSt   = (size_t)SNN_B * SNN_H * 4;          //   131,072 each
    const size_t szXq   = (size_t)SNN_B * SNN_T * SNN_I * 2;  // 16,384,000
    const size_t baseSz = szWq + szZbuf + 3 * szSt;

    char* ws = (char*)d_ws;
    __bf16* Wq   = (__bf16*)ws;
    __bf16* zbuf = (__bf16*)(ws + szWq);
    float*  v    = (float*)(ws + szWq + szZbuf);
    float*  a    = v  + SNN_B * SNN_H;
    float*  tr   = a  + SNN_B * SNN_H;
    __bf16* xq   = (__bf16*)(ws + baseSz);
    const bool use_xq = (ws_size >= baseSz + szXq);

    // Precompute bf16 effective weights (+input tensor); zero recurrent state.
    {
        const int n = SNN_H * SNN_K;
        snn_build_w<<<(n + 255) / 256, 256, 0, stream>>>(C0, C1, bw, Wq);
        snn_init_state<<<(SNN_B * SNN_H + 255) / 256, 256, 0, stream>>>(v, a, tr, zbuf);
        if (use_xq) {
            const size_t nx = (size_t)SNN_B * SNN_T * SNN_I;
            snn_build_xq<<<(unsigned)((nx + 255) / 256), 256, 0, stream>>>(x, xq);
        }
    }

    float* spikes_out = out + OFF_SPIKES;
    float* trace_out  = out + OFF_TRACES;

    // Sequential recurrence: one fused WMMA step-kernel per timestep.
    if (use_xq) {
        for (int t = 0; t < SNN_T; ++t)
            snn_step<true><<<SNN_H / 16, 64, 0, stream>>>(
                x, xq, Wq, zbuf, v, a, tr, spikes_out, trace_out, t);
    } else {
        for (int t = 0; t < SNN_T; ++t)
            snn_step<false><<<SNN_H / 16, 64, 0, stream>>>(
                x, xq, Wq, zbuf, v, a, tr, spikes_out, trace_out, t);
    }

    // Step t writes zbuf[(t+1)&1]; t=999 -> buffer 0 holds final spikes.
    snn_finalize<<<1 + (SNN_B * SNN_H + 255) / 256, 256, 0, stream>>>(
        tr, v, a, zbuf, W_out, b_out, out);
}
